// DeepseekV2Layer_79336635892597
// MI455X (gfx1250) — matrix operand order repaired and checked by reading, not compiled, not run on hardware
//
#include <hip/hip_runtime.h>
#include <hip/hip_bf16.h>

// ---------------------------------------------------------------------------
// DeepSeek-V2 MoE layer for MI455X (gfx1250), bf16 WMMA path, async-LDS B.
//   1) fp32 -> bf16 conversion of hidden + all weights (workspace)
//   2) router: logits -> softmax -> grouped top-k -> dense combine [M,E]
//   3) expert gate/up GEMM + silu*mul (comb folded) -> g_bf [E,M,N]
//   4) shared gate/up GEMM + silu*mul -> act_bf [M,ISH]
//   5) shared down GEMM (writes d_out)
//   6) expert down GEMM, grouped-accumulate over E (adds into d_out)
// GEMM blocks: 8 waves stacked on M (256x64 macro-tile); B tiles staged into
// LDS with global_load_async_to_lds_b128 (double buffered, ASYNCcnt), WMMA
// math with v_wmma_f32_16x16x32_bf16, A tiles direct global_load_b128.
// ---------------------------------------------------------------------------

typedef __attribute__((ext_vector_type(16))) __bf16 bf16x16;
typedef __attribute__((ext_vector_type(8)))  __bf16 bf16x8;
typedef __attribute__((ext_vector_type(8)))  float  v8f;

__device__ __forceinline__ unsigned short f32_to_bf16_rne(float f) {
    unsigned int u = __float_as_uint(f);
    unsigned int r = u + 0x7fffu + ((u >> 16) & 1u);
    return (unsigned short)(r >> 16);
}

// Async copy of 16B/lane from global (SGPR base + 32-bit lane offset, GVS) to
// per-lane LDS byte address. Tracked by ASYNCcnt.
__device__ __forceinline__ void async_g2l_b128(unsigned lds_byte, unsigned gvoff_byte,
                                               unsigned long long sbase) {
    asm volatile("global_load_async_to_lds_b128 %0, %1, %2"
                 :: "v"(lds_byte), "v"(gvoff_byte), "s"(sbase) : "memory");
}
__device__ __forceinline__ void wait_async0() {
    asm volatile("s_wait_asynccnt 0" ::: "memory");
}
__device__ __forceinline__ unsigned lds_off_u32(const void* p) {
    return (unsigned)(unsigned long long)p;   // low 32 bits of shared aperture
}

// ---------------------------------------------------------------------------
// 1) elementwise fp32 -> bf16
// ---------------------------------------------------------------------------
__global__ void cvt_f32_bf16_kernel(const float* __restrict__ src,
                                    unsigned short* __restrict__ dst, long n) {
    long i = (long)blockIdx.x * blockDim.x + threadIdx.x;
    long stride = (long)gridDim.x * blockDim.x;
    for (; i < n; i += stride) dst[i] = f32_to_bf16_rne(src[i]);
}

// ---------------------------------------------------------------------------
// 2) router (one thread per token, gate_w in LDS)
// ---------------------------------------------------------------------------
__global__ __launch_bounds__(256)
void router_kernel(const float* __restrict__ hidden,
                   const float* __restrict__ gate_w,
                   float* __restrict__ comb, int M, int H) {
    constexpr int E = 16, NG = 8, TOPKG = 3, TOPK = 6;
    __shared__ float sgw[E * 1024];
    const int tid = threadIdx.x;
    for (int i = tid; i < E * H; i += blockDim.x) sgw[i] = gate_w[i];
    __syncthreads();

    const int m = blockIdx.x * blockDim.x + tid;
    if (m >= M) return;

    float acc[E];
#pragma unroll
    for (int e = 0; e < E; ++e) acc[e] = 0.0f;
    const float* hrow = hidden + (long)m * H;
    for (int h = 0; h < H; ++h) {
        float x = hrow[h];
#pragma unroll
        for (int e = 0; e < E; ++e) acc[e] += x * sgw[e * 1024 + h];
    }
    float mx = acc[0];
#pragma unroll
    for (int e = 1; e < E; ++e) mx = fmaxf(mx, acc[e]);
    float sum = 0.0f;
#pragma unroll
    for (int e = 0; e < E; ++e) { acc[e] = __expf(acc[e] - mx); sum += acc[e]; }
    float inv = 1.0f / sum;
#pragma unroll
    for (int e = 0; e < E; ++e) acc[e] *= inv;
    float gs[NG];
#pragma unroll
    for (int g = 0; g < NG; ++g) gs[g] = fmaxf(acc[2 * g], acc[2 * g + 1]);
    bool gsel[NG];
#pragma unroll
    for (int g = 0; g < NG; ++g) gsel[g] = false;
    for (int t = 0; t < TOPKG; ++t) {
        int best = -1; float bv = -1e30f;
#pragma unroll
        for (int g = 0; g < NG; ++g)
            if (!gsel[g] && gs[g] > bv) { bv = gs[g]; best = g; }
        gsel[best] = true;
    }
    float ms[E], row[E];
#pragma unroll
    for (int e = 0; e < E; ++e) { ms[e] = gsel[e >> 1] ? acc[e] : 0.0f; row[e] = 0.0f; }
    for (int t = 0; t < TOPK; ++t) {
        int best = 0; float bv = -1e30f;
#pragma unroll
        for (int e = 0; e < E; ++e)
            if (ms[e] > bv) { bv = ms[e]; best = e; }
        row[best] = bv > 0.0f ? bv : 0.0f;
        ms[best] = -1e30f;
    }
#pragma unroll
    for (int e = 0; e < E; ++e) comb[(long)m * E + e] = row[e];
}

// ---------------------------------------------------------------------------
// WMMA A-tile loader (16x32 A, contiguous k order used by this code): lane L (m=L&15, kh=L>>4)
// holds K run [kh*16, kh*16+16): two 16B global loads.
// B 32x16 (same k order): lane L (n=L&15, kh=L>>4) holds run [kh*16, kh*16+16): 32B.
// ---------------------------------------------------------------------------
__device__ __forceinline__ bf16x16 load_a_tile(const unsigned short* base,
                                               long ld, int mlane, int khalf) {
    const unsigned short* p = base + (long)mlane * ld + khalf * 16;
    bf16x8 lo = *(const bf16x8*)p;
    bf16x8 hi = *(const bf16x8*)(p + 8);
    return __builtin_shufflevector(lo, hi, 0, 1, 2, 3, 4, 5, 6, 7,
                                   8, 9, 10, 11, 12, 13, 14, 15);
}

// ---------------------------------------------------------------------------
// 3/4) gate+up GEMM, fused silu*mul and optional combine scale.
//   Block: 8 waves -> 256(M) x 64(F) macro-tile, both halves.
//   B tiles (gate 64x32 + up 64x32 bf16 = 8KB/step) async-staged to LDS,
//   double buffered; each of 256 threads copies one 16B chunk per matrix.
// ---------------------------------------------------------------------------
__global__ __launch_bounds__(256)
void gemm_gateup_silu_kernel(const unsigned short* __restrict__ A,
                             const unsigned short* __restrict__ B,
                             const float* __restrict__ comb,   // [M,E] or null
                             unsigned short* __restrict__ G,
                             int M, int K, int Nout, int E) {
    __shared__ __align__(32) unsigned short sB[2][2][64 * 32]; // [buf][g/u][row*32+k]
    const int tid = threadIdx.x;
    const int lane = tid & 31;
    const int waveInBlk = tid >> 5;

    const int tilesF = Nout >> 6;
    const int blocksM = M >> 8;                 // 256-row macro blocks
    const int perE = blocksM * tilesF;
    const int e = blockIdx.x / perE;
    if (e >= E) return;
    const int rem = blockIdx.x % perE;
    const int bm = rem / tilesF, tf = rem % tilesF;
    const int mbase = (bm << 8) + (waveInBlk << 5);
    const int fbase = tf << 6;
    const int mlane = lane & 15, khalf = lane >> 4;
    const int nlane = lane & 15;

    const unsigned short* Be = B + (long)e * 2 * Nout * K;
    unsigned short* Ge = G + (long)e * M * Nout;
    const unsigned long long sbase = (unsigned long long)Be;

    // staging role of this thread: row (0..63), 16B chunk (0..3) within k-run
    const int srow = tid >> 2, sc = tid & 3;
    const unsigned gro_g = (unsigned)(((fbase + srow) * (long)K + sc * 8) * 2);
    const unsigned gro_u = (unsigned)(((Nout + fbase + srow) * (long)K + sc * 8) * 2);
    const unsigned lro   = (unsigned)((srow * 32 + sc * 8) * 2);
    const unsigned ldsG  = lds_off_u32(&sB[0][0][0]);
    const unsigned bufStride = 2u * 64 * 32 * 2;   // bytes per buffer
    const unsigned halfStride = 64u * 32 * 2;      // gate->up bytes

    auto stage = [&](int kb, int buf) {
        unsigned lb = ldsG + (unsigned)buf * bufStride + lro;
        unsigned go = (unsigned)(kb * 2);
        async_g2l_b128(lb,              gro_g + go, sbase);
        async_g2l_b128(lb + halfStride, gro_u + go, sbase);
    };

    v8f acc_g[2][4], acc_u[2][4];
#pragma unroll
    for (int mi = 0; mi < 2; ++mi)
#pragma unroll
        for (int fi = 0; fi < 4; ++fi) { acc_g[mi][fi] = (v8f)(0.0f); acc_u[mi][fi] = (v8f)(0.0f); }

    stage(0, 0);
    int cur = 0;
    for (int kb = 0; kb < K; kb += 32) {
        wait_async0();
        __syncthreads();
        if (kb + 32 < K) stage(kb + 32, cur ^ 1);

        bf16x16 a[2];
#pragma unroll
        for (int mi = 0; mi < 2; ++mi)
            a[mi] = load_a_tile(A + (long)(mbase + mi * 16) * K + kb, K, mlane, khalf);
        __builtin_prefetch(A + (long)(mbase + mlane) * K + kb + 64, 0, 1);
#pragma unroll
        for (int fi = 0; fi < 4; ++fi) {
            bf16x16 bg = *(const bf16x16*)&sB[cur][0][(fi * 16 + nlane) * 32 + khalf * 16];
            bf16x16 bu = *(const bf16x16*)&sB[cur][1][(fi * 16 + nlane) * 32 + khalf * 16];
#pragma unroll
            for (int mi = 0; mi < 2; ++mi) {
                acc_g[mi][fi] = __builtin_amdgcn_wmma_f32_16x16x32_bf16(
                    false, a[mi], false, bg, (short)0, acc_g[mi][fi], false, false);
                acc_u[mi][fi] = __builtin_amdgcn_wmma_f32_16x16x32_bf16(
                    false, a[mi], false, bu, (short)0, acc_u[mi][fi], false, false);
            }
        }
        cur ^= 1;
    }

    // epilogue: g = comb * silu(h_gate) * h_up -> bf16
    // C/D layout: VGPR r, lane L -> (m = khalf*8 + r, n = L&15) within tile.
#pragma unroll
    for (int mi = 0; mi < 2; ++mi) {
        float w[8];
#pragma unroll
        for (int r = 0; r < 8; ++r) {
            int m = mbase + mi * 16 + khalf * 8 + r;
            w[r] = comb ? comb[(long)m * E + e] : 1.0f;
        }
#pragma unroll
        for (int fi = 0; fi < 4; ++fi)
#pragma unroll
            for (int r = 0; r < 8; ++r) {
                float gg = acc_g[mi][fi][r];
                float uu = acc_u[mi][fi][r];
                float s = (gg / (1.0f + __expf(-gg))) * uu * w[r];
                int m = mbase + mi * 16 + khalf * 8 + r;
                int n = fbase + fi * 16 + nlane;
                Ge[(long)m * Nout + n] = f32_to_bf16_rne(s);
            }
    }
}

// ---------------------------------------------------------------------------
// 5/6) down GEMM, grouped-accumulate over experts:
//   Out[m,h] (+)= scale * sum_e sum_n G[e,m,n] * W[e,h,n]
//   Block: 8 waves -> 256(M) x 64(H); W tiles (4KB/step) async-staged to LDS.
// ---------------------------------------------------------------------------
__global__ __launch_bounds__(256)
void gemm_down_kernel(const unsigned short* __restrict__ Gm,  // [E,M,N]
                      const unsigned short* __restrict__ W,   // [E,H,N]
                      float* __restrict__ Out,                // [M,H]
                      int M, int N, int H, int E, int addOut, float scale) {
    __shared__ __align__(32) unsigned short sW[2][64 * 32];
    const int tid = threadIdx.x;
    const int lane = tid & 31;
    const int waveInBlk = tid >> 5;

    const int tilesH = H >> 6;
    const int bm = blockIdx.x / tilesH, th = blockIdx.x % tilesH;
    if (bm >= (M >> 8)) return;
    const int mbase = (bm << 8) + (waveInBlk << 5);
    const int hbase = th << 6;
    const int mlane = lane & 15, khalf = lane >> 4;
    const int nlane = lane & 15;

    const int srow = tid >> 2, sc = tid & 3;
    const unsigned lro  = (unsigned)((srow * 32 + sc * 8) * 2);
    const unsigned ldsW = lds_off_u32(&sW[0][0]);
    const unsigned bufStride = 64u * 32 * 2;
    const int stepsPerE = N >> 5;
    const int S = E * stepsPerE;

    auto stage = [&](int s, int buf) {
        int e = s / stepsPerE;
        int nb = (s - e * stepsPerE) << 5;
        unsigned long long sb = (unsigned long long)(W + (long)e * H * N);
        unsigned go = (unsigned)(((hbase + srow) * (long)N + nb + sc * 8) * 2);
        async_g2l_b128(ldsW + (unsigned)buf * bufStride + lro, go, sb);
    };

    v8f acc[2][4];
#pragma unroll
    for (int mi = 0; mi < 2; ++mi)
#pragma unroll
        for (int hi = 0; hi < 4; ++hi) acc[mi][hi] = (v8f)(0.0f);

    stage(0, 0);
    for (int s = 0; s < S; ++s) {
        wait_async0();
        __syncthreads();
        if (s + 1 < S) stage(s + 1, (s + 1) & 1);

        int e = s / stepsPerE;
        int nb = (s - e * stepsPerE) << 5;
        const unsigned short* Ge = Gm + (long)e * M * N;
        const int cur = s & 1;

        bf16x16 a[2];
#pragma unroll
        for (int mi = 0; mi < 2; ++mi)
            a[mi] = load_a_tile(Ge + (long)(mbase + mi * 16) * N + nb, N, mlane, khalf);
        __builtin_prefetch(Ge + (long)(mbase + mlane) * N + nb + 64, 0, 1);
#pragma unroll
        for (int hi = 0; hi < 4; ++hi) {
            bf16x16 b = *(const bf16x16*)&sW[cur][(hi * 16 + nlane) * 32 + khalf * 16];
#pragma unroll
            for (int mi = 0; mi < 2; ++mi)
                acc[mi][hi] = __builtin_amdgcn_wmma_f32_16x16x32_bf16(
                    false, a[mi], false, b, (short)0, acc[mi][hi], false, false);
        }
    }

#pragma unroll
    for (int mi = 0; mi < 2; ++mi)
#pragma unroll
        for (int hi = 0; hi < 4; ++hi)
#pragma unroll
            for (int r = 0; r < 8; ++r) {
                int m = mbase + mi * 16 + khalf * 8 + r;
                int h = hbase + hi * 16 + nlane;
                float v = acc[mi][hi][r] * scale;
                long idx = (long)m * H + h;
                if (addOut) Out[idx] += v; else Out[idx] = v;
            }
}

// ---------------------------------------------------------------------------
// Launcher
// ---------------------------------------------------------------------------
extern "C" void kernel_launch(void* const* d_in, const int* in_sizes, int n_in,
                              void* d_out, int out_size, void* d_ws, size_t ws_size,
                              hipStream_t stream) {
    (void)in_sizes; (void)n_in; (void)out_size; (void)ws_size;
    const float* hidden = (const float*)d_in[0];
    const float* gate_w = (const float*)d_in[1];
    const float* w1     = (const float*)d_in[2];
    const float* w2     = (const float*)d_in[3];
    const float* sgu    = (const float*)d_in[4];
    const float* sd     = (const float*)d_in[5];
    float* out = (float*)d_out;

    const int M = 1024, H = 1024, Nm = 1024, E = 16, ISH = 2048;

    char* ws = (char*)d_ws;
    size_t off = 0;
    auto take = [&](size_t bytes) { char* p = ws + off; off += (bytes + 255) & ~(size_t)255; return p; };
    unsigned short* hid_bf = (unsigned short*)take((size_t)M * H * 2);
    unsigned short* w1_bf  = (unsigned short*)take((size_t)E * 2 * Nm * H * 2);
    unsigned short* w2_bf  = (unsigned short*)take((size_t)E * H * Nm * 2);
    unsigned short* sgu_bf = (unsigned short*)take((size_t)2 * ISH * H * 2);
    unsigned short* sd_bf  = (unsigned short*)take((size_t)H * ISH * 2);
    unsigned short* g_bf   = (unsigned short*)take((size_t)E * M * Nm * 2);
    unsigned short* act_bf = (unsigned short*)take((size_t)M * ISH * 2);
    float*          comb   = (float*)take((size_t)M * E * 4);

    auto cvt = [&](const float* s, unsigned short* d, long n) {
        int blk = 256;
        long g = (n + blk - 1) / blk;
        int grid = (int)(g > 4096 ? 4096 : g);
        cvt_f32_bf16_kernel<<<grid, blk, 0, stream>>>(s, d, n);
    };
    cvt(hidden, hid_bf, (long)M * H);
    cvt(w1,     w1_bf,  (long)E * 2 * Nm * H);
    cvt(w2,     w2_bf,  (long)E * H * Nm);
    cvt(sgu,    sgu_bf, (long)2 * ISH * H);
    cvt(sd,     sd_bf,  (long)H * ISH);

    router_kernel<<<M / 256, 256, 0, stream>>>(hidden, gate_w, comb, M, H);

    // expert gate/up + silu*mul (comb folded in): E * (M/256) * (Nm/64) blocks
    gemm_gateup_silu_kernel<<<E * (M / 256) * (Nm / 64), 256, 0, stream>>>(
        hid_bf, w1_bf, comb, g_bf, M, H, Nm, E);
    // shared gate/up + silu*mul: (M/256) * (ISH/64) blocks
    gemm_gateup_silu_kernel<<<(M / 256) * (ISH / 64), 256, 0, stream>>>(
        hid_bf, sgu_bf, nullptr, act_bf, M, H, ISH, 1);
    // shared down: writes d_out
    gemm_down_kernel<<<(M / 256) * (H / 64), 256, 0, stream>>>(
        act_bf, sd_bf, out, M, ISH, H, 1, /*add=*/0, 1.0f);
    // routed down: grouped-accumulate over all 16 experts into d_out
    gemm_down_kernel<<<(M / 256) * (H / 64), 256, 0, stream>>>(
        g_bf, w2_bf, out, M, Nm, H, E, /*add=*/1, 1.0f);
}
